// BranchNet1d_attention_73615739453984
// MI455X (gfx1250) — matrix operand order repaired and checked by reading, not compiled, not run on hardware
//
#include <hip/hip_runtime.h>
#include <hip/hip_bf16.h>
#include <math.h>

#define B_  16
#define N_  2048
#define D_  256
#define H_  256
#define BN_ (B_ * N_)

typedef __attribute__((ext_vector_type(16))) __bf16 v16bf;
typedef __attribute__((ext_vector_type(8)))  float  v8f;

union FragA {
  v16bf  v;
  float4 f4[2];
};

__device__ __forceinline__ float gelu_exact(float x) {
  return 0.5f * x * (1.0f + erff(x * 0.70710678118654752440f));
}

__device__ __forceinline__ v8f vzero8() {
  v8f z;
#pragma unroll
  for (int i = 0; i < 8; ++i) z[i] = 0.0f;
  return z;
}

// ---------------------------------------------------------------------------
// Transpose + convert f32 weight [K, Nc] row-major -> bf16 Wt [Nc, K]
// ---------------------------------------------------------------------------
__global__ void k_transpose_w(const float* __restrict__ W, __bf16* __restrict__ Wt,
                              int K, int Nc) {
  int idx = blockIdx.x * blockDim.x + threadIdx.x;
  if (idx >= K * Nc) return;
  int k = idx / Nc;
  int n = idx % Nc;
  Wt[(size_t)n * K + k] = (__bf16)W[idx];
}

// ---------------------------------------------------------------------------
// act1[bn, j] = gelu(x[bn]*W1a[0,j] + grid[n]*W1a[1,j] + b1a[j]), bf16
// ---------------------------------------------------------------------------
__global__ void k_fnn1_act(const float* __restrict__ x, const float* __restrict__ grid,
                           const float* __restrict__ W1a, const float* __restrict__ b1a,
                           __bf16* __restrict__ act) {
  int idx = blockIdx.x * blockDim.x + threadIdx.x;
  if (idx >= BN_ * H_) return;
  int j  = idx % H_;
  int bn = idx / H_;
  int n  = bn % N_;
  float v = x[bn] * W1a[j] + grid[n] * W1a[H_ + j] + b1a[j];
  act[idx] = (__bf16)gelu_exact(v);
}

// ---------------------------------------------------------------------------
// Generic bf16 WMMA GEMM: C[M,Nc] = op(A[M,K] @ Wt^T + bias)
//   A  : bf16 row-major, leading dim lda
//   Wt : bf16 [Nc, K] (pre-transposed weights)
//   out: bf16 (Cb) or f32 (Cf), row-major ldc
// One wave computes a 16 (M) x 128 (Nc) tile (8 accumulators = 64 VGPRs).
// ---------------------------------------------------------------------------
template <bool DO_GELU>
__global__ void k_gemm_bf16(const __bf16* __restrict__ A, int lda,
                            const __bf16* __restrict__ Wt,
                            const float* __restrict__ bias,
                            __bf16* __restrict__ Cb, float* __restrict__ Cf, int ldc,
                            int M, int K, int Nc) {
  int gtid = blockIdx.x * blockDim.x + threadIdx.x;
  int wave = gtid >> 5;
  int lane = threadIdx.x & 31;
  int nGroups = Nc >> 7;            // 128-wide column groups
  int mb = wave / nGroups;
  int cg = wave % nGroups;
  if (mb >= (M >> 4)) return;       // wave-uniform
  int m0 = mb << 4;
  int c0 = cg << 7;
  bool hi = lane >= 16;
  int  lm = lane & 15;

  v8f acc[8];
#pragma unroll
  for (int t = 0; t < 8; ++t) acc[t] = vzero8();

  for (int k0 = 0; k0 < K; k0 += 32) {
    FragA af;
    const __bf16* arow = A + (size_t)(m0 + lm) * lda + k0 + (hi ? 8 : 0);
    af.f4[0] = *(const float4*)(arow);
    af.f4[1] = *(const float4*)(arow + 16);
#pragma unroll
    for (int t = 0; t < 8; ++t) {
      FragA bf;
      const __bf16* brow = Wt + (size_t)(c0 + t * 16 + lm) * K + k0 + (hi ? 8 : 0);
      bf.f4[0] = *(const float4*)(brow);
      bf.f4[1] = *(const float4*)(brow + 16);
      acc[t] = __builtin_amdgcn_wmma_f32_16x16x32_bf16(
          false, af.v, false, bf.v, (short)0, acc[t], false, false);
    }
  }

#pragma unroll
  for (int t = 0; t < 8; ++t) {
    int col = c0 + t * 16 + lm;
    float bv = bias ? bias[col] : 0.0f;
#pragma unroll
    for (int v = 0; v < 8; ++v) {
      int row = m0 + v + (hi ? 8 : 0);
      float val = acc[t][v] + bv;
      if (DO_GELU) val = gelu_exact(val);
      if (Cb) Cb[(size_t)row * ldc + col] = (__bf16)val;
      else    Cf[(size_t)row * ldc + col] = val;
    }
  }
}

// ---------------------------------------------------------------------------
// Vt[b, d, n] = qkv[(b*N + n)*768 + 512 + d]
// ---------------------------------------------------------------------------
__global__ void k_transpose_v(const __bf16* __restrict__ qkv, __bf16* __restrict__ Vt) {
  int idx = blockIdx.x * blockDim.x + threadIdx.x;
  if (idx >= B_ * N_ * D_) return;
  int d = idx % D_;
  int n = (idx / D_) % N_;
  int b = idx / (D_ * N_);
  Vt[((size_t)b * D_ + d) * N_ + n] = qkv[((size_t)b * N_ + n) * (3 * D_) + 2 * D_ + d];
}

// ---------------------------------------------------------------------------
// Flash attention, cooperative wave-pair version.
// blockDim = 128 (4 waves = 2 pairs). Each pair owns one 16-query block.
// Within a pair: wave computes partial S over half of d (no duplicated WMMA),
// halves are summed via LDS; each wave accumulates O for 128 output columns
// (8 v8f = 64 VGPRs), eliminating the register spills of the 256-col version.
// grid = B * (N/16) / 2 blocks.
// ---------------------------------------------------------------------------
__global__ void __launch_bounds__(128, 1)
k_attention(const __bf16* __restrict__ qkv,
            const __bf16* __restrict__ Vt,
            __bf16* __restrict__ O) {
  __shared__ __align__(16) float  sPart[2][2][16 * 32];  // [pair][half] partial S
  __shared__ __align__(16) __bf16 pTile[2][16 * 32];     // [pair] P (bf16)

  int widx = threadIdx.x >> 5;  // 0..3
  int lane = threadIdx.x & 31;
  int pair = widx >> 1;         // 0..1
  int half = widx & 1;          // 0..1
  bool hi = lane >= 16;
  int  lm = lane & 15;

  int qblk = blockIdx.x * 2 + pair;   // global 16-row query block id
  int qb = qblk % (N_ / 16);
  int b  = qblk / (N_ / 16);
  int m0 = qb * 16;

  const __bf16* Qb = qkv + (size_t)b * N_ * (3 * D_);
  const __bf16* Kb = Qb + D_;
  const __bf16* Vb = Vt + (size_t)b * D_ * N_;
  const int dbase = half * 128;   // this wave's K-dim (feature) half for S
  const int c0    = half * 128;   // this wave's output column base for O

  float rowmax[8], rowsum[8];
#pragma unroll
  for (int v = 0; v < 8; ++v) { rowmax[v] = -1e30f; rowsum[v] = 0.0f; }
  v8f acc[8];
#pragma unroll
  for (int t = 0; t < 8; ++t) acc[t] = vzero8();

  const float scale = 0.0625f;  // D^-0.5 = 1/16
  float* myS = sPart[pair][half];
  const float* otS = sPart[pair][1 - half];
  __bf16* myP = pTile[pair];

  for (int kk = 0; kk < N_; kk += 32) {
    // ---- partial S = Q[:, dbase:dbase+128] @ K^T over 32 keys ----
    v8f s0 = vzero8(), s1 = vzero8();
    for (int dd = 0; dd < 128; dd += 32) {
      int d0 = dbase + dd;
      FragA af;
      const __bf16* qrow = Qb + (size_t)(m0 + lm) * (3 * D_) + d0 + (hi ? 8 : 0);
      af.f4[0] = *(const float4*)(qrow);
      af.f4[1] = *(const float4*)(qrow + 16);
      FragA bk0, bk1;
      const __bf16* kr0 = Kb + (size_t)(kk + lm) * (3 * D_) + d0 + (hi ? 8 : 0);
      bk0.f4[0] = *(const float4*)(kr0);
      bk0.f4[1] = *(const float4*)(kr0 + 16);
      const __bf16* kr1 = Kb + (size_t)(kk + 16 + lm) * (3 * D_) + d0 + (hi ? 8 : 0);
      bk1.f4[0] = *(const float4*)(kr1);
      bk1.f4[1] = *(const float4*)(kr1 + 16);
      s0 = __builtin_amdgcn_wmma_f32_16x16x32_bf16(false, af.v, false, bk0.v, (short)0, s0, false, false);
      s1 = __builtin_amdgcn_wmma_f32_16x16x32_bf16(false, af.v, false, bk1.v, (short)0, s1, false, false);
    }

    // ---- publish partial S, combine halves ----
#pragma unroll
    for (int v = 0; v < 8; ++v) {
      int row = v + (hi ? 8 : 0);
      myS[row * 32 + lm]      = s0[v];
      myS[row * 32 + 16 + lm] = s1[v];
    }
    __syncthreads();

    // ---- online softmax (both waves of the pair compute identical values) ----
    float alpha[8];
#pragma unroll
    for (int v = 0; v < 8; ++v) {
      int row = v + (hi ? 8 : 0);
      float a = (s0[v] + otS[row * 32 + lm])      * scale;
      float c = (s1[v] + otS[row * 32 + 16 + lm]) * scale;
      float mx = fmaxf(a, c);
#pragma unroll
      for (int off = 1; off < 16; off <<= 1)
        mx = fmaxf(mx, __shfl_xor(mx, off, 32));
      float nm = fmaxf(rowmax[v], mx);
      alpha[v] = __expf(rowmax[v] - nm);
      rowmax[v] = nm;
      float p0 = __expf(a - nm);
      float p1 = __expf(c - nm);
      s0[v] = p0;
      s1[v] = p1;
      float ps = p0 + p1;
#pragma unroll
      for (int off = 1; off < 16; off <<= 1)
        ps += __shfl_xor(ps, off, 32);
      rowsum[v] = rowsum[v] * alpha[v] + ps;
    }
#pragma unroll
    for (int t = 0; t < 8; ++t)
#pragma unroll
      for (int v = 0; v < 8; ++v) acc[t][v] *= alpha[v];

    // ---- P -> LDS (one wave of the pair writes; values are identical) ----
    if (half == 0) {
#pragma unroll
      for (int v = 0; v < 8; ++v) {
        int row = v + (hi ? 8 : 0);
        myP[row * 32 + lm]      = (__bf16)s0[v];
        myP[row * 32 + 16 + lm] = (__bf16)s1[v];
      }
    }
    __syncthreads();

    // ---- O[:, c0:c0+128] += P @ V ----
    FragA pf;
    const __bf16* prow = myP + lm * 32 + (hi ? 8 : 0);
    pf.f4[0] = *(const float4*)(prow);
    pf.f4[1] = *(const float4*)(prow + 16);
#pragma unroll
    for (int t = 0; t < 8; ++t) {
      FragA vf;
      const __bf16* vrow = Vb + (size_t)(c0 + t * 16 + lm) * N_ + kk + (hi ? 8 : 0);
      vf.f4[0] = *(const float4*)(vrow);
      vf.f4[1] = *(const float4*)(vrow + 16);
      acc[t] = __builtin_amdgcn_wmma_f32_16x16x32_bf16(
          false, pf.v, false, vf.v, (short)0, acc[t], false, false);
    }
  }

  // ---- normalize and store this wave's 128 columns ----
#pragma unroll
  for (int v = 0; v < 8; ++v) rowsum[v] = 1.0f / rowsum[v];
#pragma unroll
  for (int t = 0; t < 8; ++t) {
#pragma unroll
    for (int v = 0; v < 8; ++v) {
      int row = m0 + v + (hi ? 8 : 0);
      int col = c0 + t * 16 + lm;
      O[((size_t)b * N_ + row) * D_ + col] = (__bf16)(acc[t][v] * rowsum[v]);
    }
  }
}

// ---------------------------------------------------------------------------
// out[b, d] = mean_n y[b, n, d]
// ---------------------------------------------------------------------------
__global__ void k_mean(const float* __restrict__ y, float* __restrict__ out) {
  int idx = blockIdx.x * blockDim.x + threadIdx.x;
  if (idx >= B_ * D_) return;
  int d = idx % D_;
  int b = idx / D_;
  float s = 0.0f;
  for (int n = 0; n < N_; ++n) s += y[((size_t)b * N_ + n) * D_ + d];
  out[idx] = s * (1.0f / N_);
}

// ---------------------------------------------------------------------------
extern "C" void kernel_launch(void* const* d_in, const int* in_sizes, int n_in,
                              void* d_out, int out_size, void* d_ws, size_t ws_size,
                              hipStream_t stream) {
  const float* x    = (const float*)d_in[0];
  const float* grid = (const float*)d_in[1];
  const float* W1a  = (const float*)d_in[2];
  const float* b1a  = (const float*)d_in[3];
  const float* W1b  = (const float*)d_in[4];
  const float* b1b  = (const float*)d_in[5];
  const float* Wqkv = (const float*)d_in[6];
  const float* W2a  = (const float*)d_in[7];
  const float* b2a  = (const float*)d_in[8];
  const float* W2b  = (const float*)d_in[9];
  const float* b2b  = (const float*)d_in[10];
  float* out = (float*)d_out;

  char* ws = (char*)d_ws;
  const size_t MB = 1024ull * 1024ull;
  // weights (bf16, transposed): 768 KB total
  __bf16* W1b_t  = (__bf16*)(ws + 0);            // [D, H]
  __bf16* Wqkv_t = (__bf16*)(ws + 128 * 1024);   // [3D, D]
  __bf16* W2a_t  = (__bf16*)(ws + 512 * 1024);   // [H, D]
  __bf16* W2b_t  = (__bf16*)(ws + 640 * 1024);   // [D, H]
  // activations (aliased where lifetimes are disjoint)
  __bf16* act1  = (__bf16*)(ws + 1 * MB);        // [BN, H] 16 MB
  __bf16* o_buf = (__bf16*)(ws + 1 * MB);        // alias act1 (dead after h-GEMM)
  __bf16* h_buf = (__bf16*)(ws + 17 * MB);       // [BN, D] 16 MB
  __bf16* act2  = (__bf16*)(ws + 17 * MB);       // alias h (dead after qkv-GEMM)
  __bf16* qkv   = (__bf16*)(ws + 33 * MB);       // [BN, 3D] 48 MB
  float*  y_buf = (float*)(ws + 33 * MB);        // alias qkv (dead after attention)
  __bf16* Vt    = (__bf16*)(ws + 81 * MB);       // [B, D, N] 16 MB

  // 1) weight transpose/convert
  k_transpose_w<<<(H_ * D_) / 256, 256, 0, stream>>>(W1b, W1b_t, H_, D_);
  k_transpose_w<<<(D_ * 3 * D_) / 256, 256, 0, stream>>>(Wqkv, Wqkv_t, D_, 3 * D_);
  k_transpose_w<<<(D_ * H_) / 256, 256, 0, stream>>>(W2a, W2a_t, D_, H_);
  k_transpose_w<<<(H_ * D_) / 256, 256, 0, stream>>>(W2b, W2b_t, H_, D_);

  // 2) act1 = gelu([x, grid] @ W1a + b1a)
  k_fnn1_act<<<(BN_ * H_) / 256, 256, 0, stream>>>(x, grid, W1a, b1a, act1);

  // 3) h = act1 @ W1b + b1b   (bf16 out)
  k_gemm_bf16<false><<<(BN_ / 16) * (D_ / 128) / 8, 256, 0, stream>>>(
      act1, H_, W1b_t, b1b, h_buf, nullptr, D_, BN_, H_, D_);

  // 4) qkv = h @ Wqkv          (bf16 out, no bias)
  k_gemm_bf16<false><<<(BN_ / 16) * ((3 * D_) / 128) / 8, 256, 0, stream>>>(
      h_buf, D_, Wqkv_t, nullptr, qkv, nullptr, 3 * D_, BN_, D_, 3 * D_);

  // 5) Vt = transpose(V)
  k_transpose_v<<<(B_ * N_ * D_) / 256, 256, 0, stream>>>(qkv, Vt);

  // 6) flash attention -> o_buf   (2 query blocks per 128-thread block)
  k_attention<<<(B_ * (N_ / 16)) / 2, 128, 0, stream>>>(qkv, Vt, o_buf);

  // 7) act2 = gelu(o @ W2a + b2a)
  k_gemm_bf16<true><<<(BN_ / 16) * (H_ / 128) / 8, 256, 0, stream>>>(
      o_buf, D_, W2a_t, b2a, act2, nullptr, H_, BN_, D_, H_);

  // 8) y = act2 @ W2b + b2b   (f32 out)
  k_gemm_bf16<false><<<(BN_ / 16) * (D_ / 128) / 8, 256, 0, stream>>>(
      act2, H_, W2b_t, b2b, nullptr, y_buf, D_, BN_, H_, D_);

  // 9) out = mean over N
  k_mean<<<(B_ * D_) / 256, 256, 0, stream>>>(y_buf, out);
}
